// LSTMFeatureExtractor_6236292514434
// MI455X (gfx1250) — matrix operand order repaired and checked by reading, not compiled, not run on hardware
//
#include <hip/hip_runtime.h>
#include <math.h>

typedef _Float16 half8 __attribute__((ext_vector_type(8)));
typedef _Float16 v16h  __attribute__((ext_vector_type(16)));
typedef float    v8f   __attribute__((ext_vector_type(8)));

namespace {
constexpr int kH  = 64;    // hidden
constexpr int kG  = 256;   // 4*H gates
constexpr int kT  = 512;   // timesteps
constexpr int kBT = 16;    // batch tile per workgroup
constexpr int kNT = 16;    // gate tiles of 16 (kG/16)
constexpr int kKC = 2;     // K chunks of 32 (kH/32)
constexpr int kOUT = 32;
constexpr int kPacked = kNT * kKC * 32 * 16; // fragment-packed halves per matrix

struct SharedMem {
  float    xbuf[kBT * kT];      // 32 KB: x tile preload
  _Float16 whh0[kPacked];       // 32 KB: B fragments, layer0 recurrent
  _Float16 wih1[kPacked];       // 32 KB: B fragments, layer1 input
  _Float16 whh1[kPacked];       // 32 KB: B fragments, layer1 recurrent
  _Float16 h1[kBT * kH];        // layer0 hidden state (f16 for WMMA A)
  _Float16 h2[kBT * kH];        // layer1 hidden state (f16 for WMMA A)
  float    h2f[kBT * kH];       // layer1 hidden state (f32 for FC)
  float    c1[kBT * kH];
  float    c2[kBT * kH];
  float    gatesA[kBT * kG];    // 16 KB: layer0 gate staging (pipelined)
  float    gatesB[kBT * kG];    // 16 KB: layer1 gate staging
  float    bsum0[kG];
  float    bsum1[kG];
  float    wih0c[kG];           // Wih0 column (IN == 1)
};
} // namespace

__device__ inline float rcp_(float x) { return __builtin_amdgcn_rcpf(x); }
__device__ inline float sigm_(float x) { return rcp_(1.0f + __expf(-x)); }
__device__ inline float tanh_(float x) {
  x = fminf(fmaxf(x, -15.0f), 15.0f);
  return 1.0f - 2.0f * rcp_(__expf(2.0f * x) + 1.0f);
}

__device__ inline v8f wmma_f16(v16h a, v16h b, v8f c) {
  // D = A(16x32,f16) * B(32x16,f16) + C(16x16,f32)
  return __builtin_amdgcn_wmma_f32_16x16x32_f16(false, a, false, b, (short)0, c,
                                                false, false);
}

// A-matrix fragment (16x32 f16) from row-major LDS [16][64], K-chunk kc.
__device__ inline v16h load_a(const _Float16* hbuf, int lane, int kc) {
  const int m = lane & 15;
  const int base = kc * 32 + (lane >> 4) * 8;
  half8 lo = *(const half8*)(hbuf + m * kH + base);
  half8 hi = *(const half8*)(hbuf + m * kH + base + 16);
  return __builtin_shufflevector(lo, hi, 0, 1, 2, 3, 4, 5, 6, 7,
                                 8, 9, 10, 11, 12, 13, 14, 15);
}

// B fragments are pre-packed lane-contiguous: one aligned 32B LDS read.
__device__ inline v16h load_b(const _Float16* packed, int nt, int kc, int lane) {
  return *(const v16h*)(packed + ((nt * kKC + kc) * 32 + lane) * 16);
}

// Pack W (256x64 row-major f32, B[k][n] = W[n][k]) into WMMA B-fragment order:
// lane l holds col n = nt*16 + (l&15); elem e -> K = kc*32 + (l>>4)*8 + (e&7) + (e>>3)*16.
__device__ inline void pack_weights(_Float16* dst, const float* W, int tid) {
  for (int p = tid; p < kPacked; p += 256) {
    const int e    = p & 15;
    const int lane = (p >> 4) & 31;
    const int tt   = p >> 9;
    const int nt   = tt >> 1;
    const int kc   = tt & 1;
    const int n    = nt * 16 + (lane & 15);
    const int k    = kc * 32 + (lane >> 4) * 8 + (e & 7) + (e >> 3) * 16;
    dst[p] = (_Float16)W[n * kH + k];
  }
}

__device__ inline void store_frag(float* gates, v8f acc, int nt, int lane) {
  const int col = nt * 16 + (lane & 15);
  const int rb  = (lane >> 4) * 8;
#pragma unroll
  for (int v = 0; v < 8; ++v) gates[(rb + v) * kG + col] = acc[v];
}

// gates(16x256) -> c,h update. Gate order i,f,g,o (64 each).
__device__ inline void lstm_ew(const float* gates, float* c, _Float16* hf,
                               float* hfull, int tid) {
#pragma unroll
  for (int u = 0; u < (kBT * kH) / 256; ++u) {
    const int idx = u * 256 + tid;
    const int m = idx >> 6, k = idx & 63;
    const float* gr = gates + m * kG;
    const float gi = sigm_(gr[k]);
    const float gf = sigm_(gr[64 + k]);
    const float gg = tanh_(gr[128 + k]);
    const float go = sigm_(gr[192 + k]);
    const float cn = fmaf(gf, c[idx], gi * gg);
    c[idx] = cn;
    const float h = go * tanh_(cn);
    hf[idx] = (_Float16)h;
    if (hfull) hfull[idx] = h;
  }
}

__global__ __launch_bounds__(256)
void lstm_fused_kernel(const float* __restrict__ x,
                       const float* __restrict__ Wih0,
                       const float* __restrict__ Whh0,
                       const float* __restrict__ bih0,
                       const float* __restrict__ bhh0,
                       const float* __restrict__ Wih1,
                       const float* __restrict__ Whh1,
                       const float* __restrict__ bih1,
                       const float* __restrict__ bhh1,
                       const float* __restrict__ fcW,
                       const float* __restrict__ fcb,
                       float* __restrict__ out) {
  __shared__ SharedMem s;
  const int tid  = threadIdx.x;
  const int lane = tid & 31;
  const int wave = tid >> 5;
  const int b0   = blockIdx.x * kBT;

  // ---------- one-time staging ----------
  for (int i = tid; i < kBT * kT; i += 256) s.xbuf[i] = x[b0 * kT + i];
  pack_weights(s.whh0, Whh0, tid);
  pack_weights(s.wih1, Wih1, tid);
  pack_weights(s.whh1, Whh1, tid);
  for (int n = tid; n < kG; n += 256) {
    s.bsum0[n] = bih0[n] + bhh0[n];
    s.bsum1[n] = bih1[n] + bhh1[n];
    s.wih0c[n] = Wih0[n];  // (4H,1)
  }
  for (int i = tid; i < kBT * kH; i += 256) {
    s.h1[i] = (_Float16)0.0f;
    s.h2[i] = (_Float16)0.0f;
    s.h2f[i] = 0.0f;
    s.c1[i] = 0.0f;
    s.c2[i] = 0.0f;
  }
  __syncthreads();

  // ---------- hoist weight B-fragments into VGPRs (reused 512x) ----------
  const int ntA = wave * 2, ntB = wave * 2 + 1;
  const v16h Bhh0A0 = load_b(s.whh0, ntA, 0, lane), Bhh0A1 = load_b(s.whh0, ntA, 1, lane);
  const v16h Bhh0B0 = load_b(s.whh0, ntB, 0, lane), Bhh0B1 = load_b(s.whh0, ntB, 1, lane);
  const v16h Bih1A0 = load_b(s.wih1, ntA, 0, lane), Bih1A1 = load_b(s.wih1, ntA, 1, lane);
  const v16h Bih1B0 = load_b(s.wih1, ntB, 0, lane), Bih1B1 = load_b(s.wih1, ntB, 1, lane);
  const v16h Bhh1A0 = load_b(s.whh1, ntA, 0, lane), Bhh1A1 = load_b(s.whh1, ntA, 1, lane);
  const v16h Bhh1B0 = load_b(s.whh1, ntB, 0, lane), Bhh1B1 = load_b(s.whh1, ntB, 1, lane);

  const int colA = ntA * 16 + (lane & 15);
  const int colB = ntB * 16 + (lane & 15);
  const float wA  = s.wih0c[colA], wB  = s.wih0c[colB];
  const float bA0 = s.bsum0[colA], bB0 = s.bsum0[colB];
  const float bA1 = s.bsum1[colA], bB1 = s.bsum1[colB];
  const int rowbase = (lane >> 4) * 8;

  // layer0 gemm for step t: gatesA = x_t*Wih0 + bsum0 + h1 @ Whh0^T
  auto gemm0 = [&](int t) {
    float xs[8];
#pragma unroll
    for (int v = 0; v < 8; ++v) xs[v] = s.xbuf[(rowbase + v) * kT + t];
    v8f accA, accB;
#pragma unroll
    for (int v = 0; v < 8; ++v) {
      accA[v] = fmaf(xs[v], wA, bA0);
      accB[v] = fmaf(xs[v], wB, bB0);
    }
    const v16h A0 = load_a(s.h1, lane, 0);
    const v16h A1 = load_a(s.h1, lane, 1);
    accA = wmma_f16(A0, Bhh0A0, accA);
    accA = wmma_f16(A1, Bhh0A1, accA);
    accB = wmma_f16(A0, Bhh0B0, accB);
    accB = wmma_f16(A1, Bhh0B1, accB);
    store_frag(s.gatesA, accA, ntA, lane);
    store_frag(s.gatesA, accB, ntB, lane);
  };

  // layer1 gemms: gatesB = bsum1 + h1_t @ Wih1^T + h2 @ Whh1^T
  auto gemm1 = [&]() {
    v8f gA, gB;
#pragma unroll
    for (int v = 0; v < 8; ++v) { gA[v] = bA1; gB[v] = bB1; }
    const v16h P0 = load_a(s.h1, lane, 0);
    const v16h P1 = load_a(s.h1, lane, 1);
    const v16h Q0 = load_a(s.h2, lane, 0);
    const v16h Q1 = load_a(s.h2, lane, 1);
    gA = wmma_f16(P0, Bih1A0, gA);
    gA = wmma_f16(P1, Bih1A1, gA);
    gA = wmma_f16(Q0, Bhh1A0, gA);
    gA = wmma_f16(Q1, Bhh1A1, gA);
    gB = wmma_f16(P0, Bih1B0, gB);
    gB = wmma_f16(P1, Bih1B1, gB);
    gB = wmma_f16(Q0, Bhh1B0, gB);
    gB = wmma_f16(Q1, Bhh1B1, gB);
    store_frag(s.gatesB, gA, ntA, lane);
    store_frag(s.gatesB, gB, ntB, lane);
  };

  // ---------- software-pipelined 2-layer recurrence ----------
  // Dependence: gemm0(t+1) needs only h1_t (ew0(t)); gemm1(t) needs h1_t, h2_{t-1}.
  // Steady state: [gemm1(t) | gemm0(t+1)]  barrier  [ew1(t) | ew0(t+1)]  barrier
  gemm0(0);
  __syncthreads();
  lstm_ew(s.gatesA, s.c1, s.h1, nullptr, tid);   // ew0(0) -> h1_0
  __syncthreads();

  for (int t = 0; t < kT - 1; ++t) {
    gemm1();        // layer1 step t
    gemm0(t + 1);   // layer0 step t+1 (reads same h1_t)
    __syncthreads();
    lstm_ew(s.gatesB, s.c2, s.h2, s.h2f, tid);   // ew1(t)   -> h2_t
    lstm_ew(s.gatesA, s.c1, s.h1, nullptr, tid); // ew0(t+1) -> h1_{t+1}
    __syncthreads();
  }
  gemm1();          // layer1 step T-1
  __syncthreads();
  lstm_ew(s.gatesB, s.c2, s.h2, s.h2f, tid);
  __syncthreads();

  // ---------- fused FC(64->32) + ReLU on final h2 ----------
  for (int idx = tid; idx < kBT * kOUT; idx += 256) {
    const int m = idx >> 5, o = idx & 31;
    float acc = fcb[o];
    const float* wr = fcW + o * kH;
    const float* hr = s.h2f + m * kH;
#pragma unroll 8
    for (int k = 0; k < kH; ++k) acc = fmaf(hr[k], wr[k], acc);
    out[(b0 + m) * kOUT + o] = fmaxf(acc, 0.0f);
  }
}

extern "C" void kernel_launch(void* const* d_in, const int* in_sizes, int n_in,
                              void* d_out, int out_size, void* d_ws, size_t ws_size,
                              hipStream_t stream) {
  (void)n_in; (void)d_ws; (void)ws_size; (void)out_size;
  const float* x    = (const float*)d_in[0];
  const float* Wih0 = (const float*)d_in[1];
  const float* Whh0 = (const float*)d_in[2];
  const float* bih0 = (const float*)d_in[3];
  const float* bhh0 = (const float*)d_in[4];
  const float* Wih1 = (const float*)d_in[5];
  const float* Whh1 = (const float*)d_in[6];
  const float* bih1 = (const float*)d_in[7];
  const float* bhh1 = (const float*)d_in[8];
  const float* fcW  = (const float*)d_in[9];
  const float* fcb  = (const float*)d_in[10];
  float* out = (float*)d_out;

  const int B = in_sizes[0] / kT;   // IN == 1
  dim3 grid(B / kBT), block(256);
  lstm_fused_kernel<<<grid, block, 0, stream>>>(x, Wih0, Whh0, bih0, bhh0,
                                                Wih1, Whh1, bih1, bhh1,
                                                fcW, fcb, out);
}
